// AttentionScaledDotProduct_29643864277600
// MI455X (gfx1250) — matrix operand order repaired and checked by reading, not compiled, not run on hardware
//
#include <hip/hip_runtime.h>

typedef __attribute__((ext_vector_type(16))) _Float16 v16h;
typedef __attribute__((ext_vector_type(8)))  _Float16 v8h;
typedef __attribute__((ext_vector_type(8)))  float    v8f;
typedef __attribute__((ext_vector_type(4)))  int      v4i;

#define D_MODEL 768
#define D_EMB   64
#define NSEQ    4096
#define NB      4

#if defined(__has_builtin)
#  if __has_builtin(__builtin_amdgcn_global_load_async_to_lds_b128)
#    define HAVE_ASYNC_LDS 1
#  endif
#endif
#ifndef HAVE_ASYNC_LDS
#  define HAVE_ASYNC_LDS 0
#endif

// ---------------------------------------------------------------------------
// Projection: out[r][e] = (sum_k X[r][k]*W[k][e] + bias[e]) * outscale, f16 out
// Grid: (B*N/16) blocks, 128 threads = 4 waves; wave w handles E-tile w.
// ---------------------------------------------------------------------------
__global__ __launch_bounds__(128)
void proj_kernel(const float* __restrict__ X, const float* __restrict__ W,
                 const float* __restrict__ bias, _Float16* __restrict__ out,
                 float outscale) {
    const int wave = threadIdx.x >> 5;      // 0..3 == E-tile
    const int lane = threadIdx.x & 31;
    const int lo   = lane & 15;
    const int hi   = lane >> 4;
    const int mtile = blockIdx.x;
    const int col   = wave * 16 + lo;

    const float* arow = X + ((size_t)mtile * 16 + lo) * D_MODEL;

    v8f acc = {};
    for (int k0 = 0; k0 < D_MODEL; k0 += 32) {
        v16h a;
        const int cbase = k0 + hi * 16;
        #pragma unroll
        for (int i = 0; i < 8; ++i) a[i]     = (_Float16)arow[cbase + i];
        #pragma unroll
        for (int i = 0; i < 8; ++i) a[8 + i] = (_Float16)arow[cbase + 8 + i];
        v16h bm;
        const int kb = k0 + hi * 16;
        #pragma unroll
        for (int e = 0; e < 16; ++e)
            bm[e] = (_Float16)W[(size_t)(kb + e) * D_EMB + col];
        acc = __builtin_amdgcn_wmma_f32_16x16x32_f16(
                  false, a, false, bm, (short)0, acc, false, false);
    }

    const float bval = bias[col];
    #pragma unroll
    for (int j = 0; j < 8; ++j) {
        const int row = j + 8 * hi;
        out[((size_t)mtile * 16 + row) * D_EMB + col] =
            (_Float16)((acc[j] + bval) * outscale);
    }
}

// ---------------------------------------------------------------------------
// Flash attention, workgroup-cooperative K/V staging:
//   - 8 waves per block, one 16-query tile per wave
//   - K tile (32x64 f16) async-copied global->LDS (double buffered)
//   - V tile stored TRANSPOSED in LDS so PV B-matrix reads are contiguous
//   - per 32-key block: 4 score WMMAs + online softmax + 4 PV WMMAs
// ---------------------------------------------------------------------------
__global__ __launch_bounds__(256)
void attn_kernel(const _Float16* __restrict__ qf, const _Float16* __restrict__ kf,
                 const _Float16* __restrict__ vf, float* __restrict__ out) {
    __shared__ _Float16 kbuf[2][32 * 64];    // [key][e]       4 KB each
    __shared__ _Float16 vtbuf[2][64 * 32];   // [e][key] (T)   4 KB each
    __shared__ _Float16 pbuf[8][16 * 32];    // per-wave P transpose staging

    const int tid  = threadIdx.x;
    const int wave = tid >> 5;
    const int lane = tid & 31;
    const int lo   = lane & 15;
    const int hi   = lane >> 4;

    const int qtile = blockIdx.x * 8 + wave;    // 0..1023
    const int b  = qtile >> 8;
    const int qt = qtile & 255;

    const _Float16* qbase = qf + (size_t)b * NSEQ * D_EMB;
    const _Float16* kbase = kf + (size_t)b * NSEQ * D_EMB;
    const _Float16* vbase = vf + (size_t)b * NSEQ * D_EMB;

    // cooperative staging indices: 256 threads x 16B cover a 32x64 f16 tile
    const int skey   = tid >> 3;            // 0..31 (key row)
    const int schunk = tid & 7;             // 0..7  (8-half chunk within row)

    auto stage = [&](int buf, int kb) {
        // K: straight copy, layout preserved
        const _Float16* kg = kbase + (size_t)(kb + skey) * D_EMB + schunk * 8;
        _Float16* kl = &kbuf[buf][skey * 64 + schunk * 8];
#if HAVE_ASYNC_LDS
        __builtin_amdgcn_global_load_async_to_lds_b128(
            (__attribute__((address_space(1))) v4i*)kg,
            (__attribute__((address_space(3))) v4i*)kl, 0, 0);
#else
        *(v8h*)kl = *(const v8h*)kg;
#endif
        // V: load row chunk, scatter transposed into LDS [e][key]
        v8h vv = *(const v8h*)(vbase + (size_t)(kb + skey) * D_EMB + schunk * 8);
        #pragma unroll
        for (int i = 0; i < 8; ++i)
            vtbuf[buf][(schunk * 8 + i) * 32 + skey] = vv[i];
    };

    // Load q A-tiles once (softmax scale pre-folded into q by projection).
    v16h qa0, qa1;
    {
        const _Float16* qrow = qbase + (size_t)(qt * 16 + lo) * D_EMB;
        const int e0 = hi * 16;
        #pragma unroll
        for (int i = 0; i < 8; ++i) { qa0[i]   = qrow[e0 + i];
                                      qa0[8+i] = qrow[e0 + 8 + i]; }
        #pragma unroll
        for (int i = 0; i < 8; ++i) { qa1[i]   = qrow[32 + e0 + i];
                                      qa1[8+i] = qrow[32 + e0 + 8 + i]; }
    }

    v8f o0 = {}, o1 = {}, o2 = {}, o3 = {};
    float m[8], l[8];
    #pragma unroll
    for (int j = 0; j < 8; ++j) { m[j] = -1e30f; l[j] = 0.0f; }

    _Float16* pw = &pbuf[wave][0];

    stage(0, 0);                             // prime the pipeline

    for (int kb = 0; kb < NSEQ; kb += 32) {
        const int buf = (kb >> 5) & 1;
        if (kb + 32 < NSEQ) {
            stage(buf ^ 1, kb + 32);         // prefetch next tile
            asm volatile("s_wait_asynccnt 0x1" ::: "memory");  // cur tile done
        } else {
            asm volatile("s_wait_asynccnt 0x0" ::: "memory");
        }
        __syncthreads();                     // all waves' staging visible

        // ---- scores S = q @ k^T (two 16-key sub-tiles, E=64 in 2 chunks) ---
        v8f s0 = {}, s1 = {};
        {
            const _Float16* krow0 = &kbuf[buf][lo * 64 + hi * 16];
            const _Float16* krow1 = &kbuf[buf][(16 + lo) * 64 + hi * 16];
            v16h k00, k01, k10, k11;
            #pragma unroll
            for (int e = 0; e < 16; ++e) {
                k00[e] = krow0[e]; k01[e] = krow0[32 + e];
                k10[e] = krow1[e]; k11[e] = krow1[32 + e];
            }
            s0 = __builtin_amdgcn_wmma_f32_16x16x32_f16(false, qa0, false, k00,
                                                        (short)0, s0, false, false);
            s0 = __builtin_amdgcn_wmma_f32_16x16x32_f16(false, qa1, false, k01,
                                                        (short)0, s0, false, false);
            s1 = __builtin_amdgcn_wmma_f32_16x16x32_f16(false, qa0, false, k10,
                                                        (short)0, s1, false, false);
            s1 = __builtin_amdgcn_wmma_f32_16x16x32_f16(false, qa1, false, k11,
                                                        (short)0, s1, false, false);
        }

        // ---- online softmax (rows live across 16 lanes of each half) -------
        v8f p0, p1;
        float alpha[8];
        #pragma unroll
        for (int j = 0; j < 8; ++j) {
            float t = fmaxf(s0[j], s1[j]);
            t = fmaxf(t, __shfl_xor(t, 1));
            t = fmaxf(t, __shfl_xor(t, 2));
            t = fmaxf(t, __shfl_xor(t, 4));
            t = fmaxf(t, __shfl_xor(t, 8));
            const float mn = fmaxf(m[j], t);
            alpha[j] = __expf(m[j] - mn);
            m[j] = mn;
            p0[j] = __expf(s0[j] - mn);
            p1[j] = __expf(s1[j] - mn);
            float r = p0[j] + p1[j];
            r += __shfl_xor(r, 1);
            r += __shfl_xor(r, 2);
            r += __shfl_xor(r, 4);
            r += __shfl_xor(r, 8);
            l[j] = l[j] * alpha[j] + r;
        }
        #pragma unroll
        for (int j = 0; j < 8; ++j) {
            o0[j] *= alpha[j]; o1[j] *= alpha[j];
            o2[j] *= alpha[j]; o3[j] *= alpha[j];
        }

        // ---- transpose P (C layout -> A layout) through per-wave LDS -------
        #pragma unroll
        for (int j = 0; j < 8; ++j) {
            const int row = j + 8 * hi;
            pw[row * 32 + lo]      = (_Float16)p0[j];
            pw[row * 32 + 16 + lo] = (_Float16)p1[j];
        }
        asm volatile("s_wait_dscnt 0x0" ::: "memory");
        v16h pa;
        {
            const _Float16* prow = pw + lo * 32 + hi * 16;
            #pragma unroll
            for (int i = 0; i < 8; ++i) { pa[i] = prow[i]; pa[8 + i] = prow[8 + i]; }
        }

        // ---- O += P @ V : V^T in LDS makes B reads contiguous --------------
        #define PV_STEP(NT, OACC)                                              \
        {                                                                      \
            v16h vbm;                                                          \
            const _Float16* vcT = &vtbuf[buf][((NT) * 16 + lo) * 32 + hi * 16];\
            _Pragma("unroll")                                                  \
            for (int e = 0; e < 16; ++e) vbm[e] = vcT[e];                      \
            OACC = __builtin_amdgcn_wmma_f32_16x16x32_f16(                     \
                       false, pa, false, vbm, (short)0, OACC, false, false);   \
        }
        PV_STEP(0, o0) PV_STEP(1, o1) PV_STEP(2, o2) PV_STEP(3, o3)
        #undef PV_STEP

        __syncthreads();                     // cur buffers free for restaging
    }

    // ---- epilogue: divide by row sums, store fp32 --------------------------
    float* obase = out + ((size_t)b * NSEQ + qt * 16) * D_EMB;
    #pragma unroll
    for (int j = 0; j < 8; ++j) {
        const int row = j + 8 * hi;
        const float inv = 1.0f / l[j];
        obase[(size_t)row * D_EMB +  0 + lo] = o0[j] * inv;
        obase[(size_t)row * D_EMB + 16 + lo] = o1[j] * inv;
        obase[(size_t)row * D_EMB + 32 + lo] = o2[j] * inv;
        obase[(size_t)row * D_EMB + 48 + lo] = o3[j] * inv;
    }
}

// ---------------------------------------------------------------------------
extern "C" void kernel_launch(void* const* d_in, const int* in_sizes, int n_in,
                              void* d_out, int out_size, void* d_ws, size_t ws_size,
                              hipStream_t stream) {
    const float* queries = (const float*)d_in[0];
    const float* keys    = (const float*)d_in[1];
    const float* values  = (const float*)d_in[2];
    const float* Wq      = (const float*)d_in[3];
    const float* bq      = (const float*)d_in[4];
    const float* Wk      = (const float*)d_in[5];
    const float* bk      = (const float*)d_in[6];
    const float* Wv      = (const float*)d_in[7];
    const float* bv      = (const float*)d_in[8];
    float* out = (float*)d_out;

    const size_t per = (size_t)NB * NSEQ * D_EMB;   // 1M f16 elems each
    _Float16* qfp = (_Float16*)d_ws;
    _Float16* kfp = qfp + per;
    _Float16* vfp = kfp + per;

    dim3 pblk(128), pgrd(NB * NSEQ / 16);           // 1024 M-tiles per matrix
    hipLaunchKernelGGL(proj_kernel, pgrd, pblk, 0, stream, queries, Wq, bq, qfp, 0.125f);
    hipLaunchKernelGGL(proj_kernel, pgrd, pblk, 0, stream, keys,    Wk, bk, kfp, 1.0f);
    hipLaunchKernelGGL(proj_kernel, pgrd, pblk, 0, stream, values,  Wv, bv, vfp, 1.0f);

    dim3 ablk(256), agrd(NB * NSEQ / 16 / 8);       // 128 blocks, 8 q-tiles each
    hipLaunchKernelGGL(attn_kernel, agrd, ablk, 0, stream, qfp, kfp, vfp, out);
}
